// PAM_1382979470118
// MI455X (gfx1250) — matrix operand order repaired and checked by reading, not compiled, not run on hardware
//
#include <hip/hip_runtime.h>
#include <hip/hip_bf16.h>
#include <math.h>

#define BN 2
#define CC 64
#define HH 256
#define WW 512
#define HWP (HH*WW)            // 131072
#define N1 (BN*CC*HH*WW)       // 16777216
#define GG 4
#define CG 16
#define WSZ 8
#define NWH (HH/WSZ)           // 32
#define NWW (WW/WSZ)           // 64
#define NWIN (BN*NWH*NWW)      // 4096
#define TOK 64

typedef __attribute__((ext_vector_type(16))) _Float16 v16h;
typedef __attribute__((ext_vector_type(8)))  float    v8f;

#if defined(__HIP_DEVICE_COMPILE__) && defined(__gfx1250__) && \
    __has_builtin(__builtin_amdgcn_global_load_async_to_lds_b128) && \
    __has_builtin(__builtin_amdgcn_s_wait_asynccnt)
#define HAVE_ASYNC_LDS 1
typedef int v4i_ __attribute__((vector_size(16)));
typedef __attribute__((address_space(1))) v4i_ gv4i;   // prints as "__device__"
typedef __attribute__((address_space(3))) v4i_ lv4i;   // prints as "__shared__"
#else
#define HAVE_ASYNC_LDS 0
#endif

// ---------------- WMMA fragment loaders (LDS, documented gfx1250 layouts) ----
// A (16x32 f16): lanes 0-15 row M=lane: halves 0-7 -> K 0..7, 8-15 -> K 16..23;
// lanes 16-31 shift K by +8.  base layout: [row][stride halves].
__device__ __forceinline__ v16h load_a_frag(const _Float16* base, int row, int stride,
                                            int kbase, int lane) {
  const _Float16* p = base + row*stride + kbase + ((lane & 16) ? 8 : 0);
  v16h a;
  #pragma unroll
  for (int i = 0; i < 8; i++) { a[i] = p[i]; a[8+i] = p[16+i]; }
  return a;
}
// B (32x16 f16): lanes 0-15 col N=lane: halves 0-7 -> K 0..7, 8-15 -> K 16..23; lanes 16-31 shift K by +8.
// base layout: [col][stride halves along K].
__device__ __forceinline__ v16h load_b_frag(const _Float16* base, int col, int stride,
                                            int kbase, int lane) {
  const _Float16* p = base + col*stride + kbase + ((lane & 16) ? 8 : 0);
  v16h b;
  #pragma unroll
  for (int i = 0; i < 8; i++) { b[i] = p[i]; b[8+i] = p[16+i]; }
  return b;
}

// ---------------- BatchNorm statistics (per image, per channel) --------------
__global__ void bn_stats_kernel(const float* __restrict__ xl,
                                const float* __restrict__ xr,
                                float* __restrict__ stats) {
  int img = blockIdx.x >> 6, ch = blockIdx.x & 63;
  const float* x = img ? xr : xl;
  __shared__ float s_sum[256], s_sq[256];
  float sum = 0.f, sq = 0.f;
  for (int i = threadIdx.x; i < BN*HWP; i += 256) {
    int b = i / HWP, p = i - b*HWP;
    float v = x[(size_t)b*CC*HWP + (size_t)ch*HWP + p];
    sum += v; sq += v*v;
  }
  s_sum[threadIdx.x] = sum; s_sq[threadIdx.x] = sq;
  __syncthreads();
  for (int s = 128; s > 0; s >>= 1) {
    if (threadIdx.x < s) { s_sum[threadIdx.x] += s_sum[threadIdx.x+s]; s_sq[threadIdx.x] += s_sq[threadIdx.x+s]; }
    __syncthreads();
  }
  if (threadIdx.x == 0) {
    float mean = s_sum[0] / (float)(BN*HWP);
    float var  = s_sq[0]  / (float)(BN*HWP) - mean*mean;
    stats[img*128 + ch]      = mean;
    stats[img*128 + 64 + ch] = rsqrtf(var + 1e-5f);
  }
}

// ---------------- BatchNorm apply (f32 for residual + f16 for WMMA conv) -----
__global__ void bn_apply_kernel(const float* __restrict__ xl, const float* __restrict__ xr,
                                const float* __restrict__ gamma, const float* __restrict__ beta,
                                const float* __restrict__ stats,
                                float* __restrict__ yl, float* __restrict__ yr,
                                _Float16* __restrict__ ylh, _Float16* __restrict__ yrh) {
  size_t i = (size_t)blockIdx.x*256 + threadIdx.x;
  if (i >= (size_t)2*N1) return;
  int img = (i >= (size_t)N1);
  size_t j = img ? i - (size_t)N1 : i;
  int ch = (int)((j / HWP) % CC);
  const float* x = img ? xr : xl;
  float mean = stats[img*128 + ch], istd = stats[img*128 + 64 + ch];
  float y = (x[j] - mean) * istd * gamma[ch] + beta[ch];
  if (img) { yr[j] = y; yrh[j] = (_Float16)y; }
  else     { yl[j] = y; ylh[j] = (_Float16)y; }
}

// ---------------- Grouped 3x3 conv as implicit-GEMM WMMA --------------------
// Per block: one (b, g), 8x16 = 128-pixel tile.  M=128, N=16, K=144 pad 160.
#define KPAD 160
__global__ __launch_bounds__(256)
void gconv3_wmma_kernel(const _Float16* __restrict__ in, const float* __restrict__ wgt,
                        const float* __restrict__ bias, const float* __restrict__ res,
                        _Float16* __restrict__ out, int do_lrelu) {
  int g = blockIdx.z & 3, b = blockIdx.z >> 2;
  int h0 = blockIdx.y*8, w0 = blockIdx.x*16;
  __shared__ _Float16 tile[CG*10*18];     // [ci][10 rows][18 cols]
  __shared__ _Float16 A[128*KPAD];        // im2col A operand [pixel][k]
  __shared__ _Float16 Bw[CG*KPAD];        // weights [co][k]
  int tid = threadIdx.x;
  int wave = tid >> 5, lane = tid & 31;

  for (int i = tid; i < CG*10*18; i += 256) {
    int ci = i/180, r = (i%180)/18, c = i%18;
    int hh = h0 - 1 + r, ww2 = w0 - 1 + c;
    _Float16 v = (_Float16)0.f;
    if (hh >= 0 && hh < HH && ww2 >= 0 && ww2 < WW)
      v = in[(size_t)b*CC*HWP + (size_t)(g*CG+ci)*HWP + hh*WW + ww2];
    tile[i] = v;
  }
  for (int i = tid; i < CG*KPAD; i += 256) {
    int co = i/KPAD, k = i%KPAD;
    Bw[i] = (k < 144) ? (_Float16)wgt[(size_t)(g*CG+co)*144 + k] : (_Float16)0.f;
  }
  __syncthreads();
  for (int i = tid; i < 128*KPAD; i += 256) {
    int p = i/KPAD, k = i%KPAD;
    _Float16 v = (_Float16)0.f;
    if (k < 144) {
      int ci = k/9, tap = k%9;
      int py = p >> 4, px = p & 15;
      v = tile[ci*180 + (py + tap/3)*18 + (px + tap%3)];
    }
    A[i] = v;
  }
  __syncthreads();

  // wave w owns M-tile w (pixels w*16..w*16+15), single N-tile (16 out ch)
  v8f acc = {};
  #pragma unroll
  for (int kk = 0; kk < 5; kk++) {
    v16h a  = load_a_frag(A,  wave*16 + (lane & 15), KPAD, kk*32, lane);
    v16h bb = load_b_frag(Bw, (lane & 15),           KPAD, kk*32, lane);
    acc = __builtin_amdgcn_wmma_f32_16x16x32_f16(false, a, false, bb, (short)0, acc, false, false);
  }
  #pragma unroll
  for (int v = 0; v < 8; v++) {
    int p  = wave*16 + v + ((lane & 16) ? 8 : 0);
    int co = lane & 15;
    int hh = h0 + (p >> 4), ww2 = w0 + (p & 15);
    float val = acc[v] + bias[g*CG+co];
    if (do_lrelu) val = val > 0.f ? val : 0.1f*val;
    size_t o = (size_t)b*CC*HWP + (size_t)(g*CG+co)*HWP + hh*WW + ww2;
    if (res) val += res[o];
    out[o] = (_Float16)val;
  }
}

// ---------------- Grouped 1x1 QKV conv as WMMA ------------------------------
// Per block: 128 consecutive pixels; per group: M=128, N=48, K=16 pad 32.
__global__ __launch_bounds__(256)
void qkv_wmma_kernel(const _Float16* __restrict__ r, const float* __restrict__ wgt,
                     const float* __restrict__ bias,
                     _Float16* __restrict__ Qh, _Float16* __restrict__ Kh,
                     _Float16* __restrict__ Vh) {
  int b = (int)(((size_t)blockIdx.x*128) / HWP);
  int hw0 = (int)(((size_t)blockIdx.x*128) % HWP);
  __shared__ _Float16 A[128*32];          // [pixel][k]
  __shared__ _Float16 Bw[48*32];          // [j][k]
  int tid = threadIdx.x;
  int wave = tid >> 5, lane = tid & 31;

  for (int g = 0; g < GG; g++) {
    if (g < GG-1)
      __builtin_prefetch(&r[(size_t)b*CC*HWP + (size_t)((g+1)*CG)*HWP + hw0 + tid], 0, 1);
    for (int i = tid; i < 128*32; i += 256) {
      int k = i >> 7, p = i & 127;        // p fastest within blocks of 128 -> coalesced
      A[p*32 + k] = (k < 16) ? r[(size_t)b*CC*HWP + (size_t)(g*CG+k)*HWP + hw0 + p]
                             : (_Float16)0.f;
    }
    for (int i = tid; i < 48*32; i += 256) {
      int j = i >> 5, k = i & 31;
      Bw[i] = (k < 16) ? (_Float16)wgt[(size_t)(g*48+j)*16 + k] : (_Float16)0.f;
    }
    __syncthreads();

    #pragma unroll
    for (int tn = 0; tn < 3; tn++) {
      v8f acc = {};
      v16h a  = load_a_frag(A,  wave*16 + (lane & 15), 32, 0, lane);
      v16h bb = load_b_frag(Bw, tn*16 + (lane & 15),   32, 0, lane);
      acc = __builtin_amdgcn_wmma_f32_16x16x32_f16(false, a, false, bb, (short)0, acc, false, false);
      #pragma unroll
      for (int v = 0; v < 8; v++) {
        int p = wave*16 + v + ((lane & 16) ? 8 : 0);
        int j = tn*16 + (lane & 15);
        int oc = g*48 + j;
        _Float16 hv = (_Float16)(acc[v] + bias[oc]);
        size_t base = (size_t)b*CC*HWP + (size_t)hw0 + p;
        if (oc < 64)        Qh[base + (size_t)oc*HWP] = hv;
        else if (oc < 128)  Kh[base + (size_t)(oc-64)*HWP] = hv;
        else                Vh[base + (size_t)(oc-128)*HWP] = hv;
      }
    }
    __syncthreads();
  }
}

// ---------------- Attention scores + softmax (one block = one window/dir) ----
__global__ __launch_bounds__(256)
void attn_scores_kernel(const _Float16* __restrict__ Qsrc,
                        const _Float16* __restrict__ Ksrc,
                        const float* __restrict__ disp, int sgn,
                        float* __restrict__ Mout) {
  int n = blockIdx.x;
  int b = n >> 11, wh = (n >> 6) & 31, ww = n & 63;
  __shared__ _Float16 Qs[TOK*64];     // [t][c]
  __shared__ _Float16 KsT[TOK*64];    // [t'][c2]  (B operand storage: [col][k])
  __shared__ float    Sc[TOK*64];
  __shared__ float    mean8[8];
  int tid = threadIdx.x;

  for (int i = tid; i < TOK*64; i += 256) {
    int t = i >> 6, c = i & 63;
    int hh = wh*8 + (t >> 3), wp = ww*8 + (t & 7);
    Qs[i] = Qsrc[(size_t)b*CC*HWP + (size_t)c*HWP + hh*WW + wp];
  }
  for (int i = tid; i < TOK*64; i += 256) {
    int tp = i >> 6, c2 = i & 63;
    int s1 = tp >> 3, s2 = tp & 7;
    // quirky bhwc-as-bchw patchify decode
    int hi = c2*4 + (wh >> 3);
    int wi = ((wh & 7) << 6) + (s1 << 3) + (ww >> 3);
    int ci = ((ww & 7) << 3) + s2;
    float d = disp[(size_t)b*HWP + hi*WW + wi];
    int idx = wi + sgn*(int)d;
    idx = idx < 0 ? 0 : (idx > WW-1 ? WW-1 : idx);
    KsT[i] = Ksrc[(size_t)b*CC*HWP + (size_t)ci*HWP + hi*WW + idx];
  }
  __syncthreads();
  if (tid < 8) {   // mean over (s2,c2) per s1: 512 contiguous halves per group
    float s = 0.f;
    const _Float16* p = &KsT[tid*512];
    for (int i = 0; i < 512; i++) s += (float)p[i];
    mean8[tid] = s * (1.0f/512.0f);
  }
  __syncthreads();
  for (int i = tid; i < TOK*64; i += 256)
    KsT[i] = (_Float16)((float)KsT[i] - mean8[i >> 9]);
  __syncthreads();

  int wave = tid >> 5, lane = tid & 31;
  #pragma unroll
  for (int rep = 0; rep < 2; rep++) {
    int tile = wave*2 + rep;
    int tm = tile >> 2, tn = tile & 3;
    v8f acc = {};
    #pragma unroll
    for (int kk = 0; kk < 2; kk++) {
      v16h a  = load_a_frag(Qs,  tm*16 + (lane & 15), 64, kk*32, lane);
      v16h bb = load_b_frag(KsT, tn*16 + (lane & 15), 64, kk*32, lane);
      acc = __builtin_amdgcn_wmma_f32_16x16x32_f16(false, a, false, bb, (short)0, acc, false, false);
    }
    #pragma unroll
    for (int v = 0; v < 8; v++) {
      int row = tm*16 + v + ((lane & 16) ? 8 : 0);
      int col = tn*16 + (lane & 15);
      Sc[row*64 + col] = acc[v];
    }
  }
  __syncthreads();
  if (tid < 64) {
    float* rp = &Sc[tid*64];
    float mx = rp[0];
    for (int k = 1; k < 64; k++) mx = fmaxf(mx, rp[k]);
    float s = 0.f;
    for (int k = 0; k < 64; k++) { float e = __expf(rp[k]-mx); rp[k] = e; s += e; }
    float inv = 1.0f/s;
    float* op = &Mout[(size_t)n*4096 + (size_t)tid*64];
    for (int k = 0; k < 64; k++) op[k] = rp[k]*inv;
  }
}

// ---------------- Relax + mask + M@V (WMMA) + blend (one block = one window) -
__global__ __launch_bounds__(256)
void attn_out_kernel(const float* __restrict__ M0, const float* __restrict__ M1,
                     const _Float16* __restrict__ Vh_l, const _Float16* __restrict__ Vh_r,
                     const float* __restrict__ d_l, const float* __restrict__ d_r,
                     const float* __restrict__ x_l, const float* __restrict__ x_r,
                     float* __restrict__ out) {
  int n = blockIdx.x;
  int b = n >> 11, wh = (n >> 6) & 31, ww = n & 63;
  __shared__ float Ma[TOK*64];
  __shared__ float Mb[TOK*64];
  __shared__ _Float16 Mh[TOK*64];     // A operand [t][t']
  __shared__ _Float16 Vs[TOK*64];     // B operand storage [c2][t']
  __shared__ float mtok[TOK];
  int tid = threadIdx.x;
  int wave = tid >> 5, lane = tid & 31;

#if HAVE_ASYNC_LDS
  {
    // Two contiguous 16KB global->LDS copies via the CDNA5 async-LDS path.
    const float* s0 = &M0[(size_t)n*4096];
    const float* s1 = &M1[(size_t)n*4096];
    #pragma unroll
    for (int it = 0; it < 4; it++) {
      int e = it*256 + tid;   // b128 transfer id (4 floats)
      __builtin_amdgcn_global_load_async_to_lds_b128(
          (gv4i*)(s0 + (size_t)e*4), (lv4i*)&Ma[e*4], 0, 0);
      __builtin_amdgcn_global_load_async_to_lds_b128(
          (gv4i*)(s1 + (size_t)e*4), (lv4i*)&Mb[e*4], 0, 0);
    }
    __builtin_amdgcn_s_wait_asynccnt(0);
  }
#else
  for (int i = tid; i < TOK*64; i += 256) {
    Ma[i] = M0[(size_t)n*4096 + i];
    Mb[i] = M1[(size_t)n*4096 + i];
  }
#endif
  __syncthreads();

  for (int dir = 0; dir < 2; dir++) {
    const float* Mc = dir == 0 ? Ma : Mb;
    const float* Mo = dir == 0 ? Mb : Ma;
    const _Float16* Vsrc = dir == 0 ? Vh_r : Vh_l;
    const float* disp = dir == 0 ? d_l : d_r;
    int sgn = dir == 0 ? -1 : +1;
    const float* xin = dir == 0 ? x_l : x_r;
    float* outp = out + (dir ? (size_t)N1 : 0);

    for (int i = tid; i < TOK*64; i += 256) {
      Mh[i] = (_Float16)Mc[i];
      int c2 = i >> 6, tp = i & 63;
      int s1 = tp >> 3, s2 = tp & 7;
      int hi = c2*4 + (wh >> 3);
      int wi = ((wh & 7) << 6) + (s1 << 3) + (ww >> 3);
      int ci = ((ww & 7) << 3) + s2;
      float d = disp[(size_t)b*HWP + hi*WW + wi];
      int idx = wi + sgn*(int)d;
      idx = idx < 0 ? 0 : (idx > WW-1 ? WW-1 : idx);
      Vs[i] = Vsrc[(size_t)b*CC*HWP + (size_t)ci*HWP + hi*WW + idx];
    }
    if (tid < 64) {
      int j = tid;
      int hh = wh*8 + (j >> 3), wp = ww*8 + (j & 7);
      __builtin_prefetch(&xin[(size_t)b*CC*HWP + hh*WW + wp], 0, 0);
      float acc = 0.f;
      for (int k = 0; k < 64; k++) {
        float rx = Mc[j*64+k];
        if (j >= 1)  rx += Mc[(j-1)*64+k];
        if (j >= 2)  rx += Mc[(j-2)*64+k];
        if (j <= 62) rx += Mc[(j+1)*64+k];
        if (j <= 61) rx += Mc[(j+2)*64+k];
        acc += rx * Mo[k*64+j];
      }
      mtok[j] = tanhf(5.0f*acc);
    }
    __syncthreads();

    #pragma unroll
    for (int rep = 0; rep < 2; rep++) {
      int tile = wave*2 + rep;
      int tm = tile >> 2, tn = tile & 3;
      v8f acc = {};
      #pragma unroll
      for (int kk = 0; kk < 2; kk++) {
        v16h a  = load_a_frag(Mh, tm*16 + (lane & 15), 64, kk*32, lane);
        v16h bb = load_b_frag(Vs, tn*16 + (lane & 15), 64, kk*32, lane);
        acc = __builtin_amdgcn_wmma_f32_16x16x32_f16(false, a, false, bb, (short)0, acc, false, false);
      }
      #pragma unroll
      for (int v = 0; v < 8; v++) {
        int t  = tm*16 + v + ((lane & 16) ? 8 : 0);
        int c2 = tn*16 + (lane & 15);
        int hh = wh*8 + (t >> 3), wp = ww*8 + (t & 7);
        float m = mtok[t];
        size_t o = (size_t)b*CC*HWP + (size_t)c2*HWP + hh*WW + wp;
        outp[o] = xin[o]*(1.0f - m) + acc[v]*m;
      }
    }
    __syncthreads();
  }
}

// ---------------- Host-side launcher -----------------------------------------
extern "C" void kernel_launch(void* const* d_in, const int* in_sizes, int n_in,
                              void* d_out, int out_size, void* d_ws, size_t ws_size,
                              hipStream_t stream) {
  (void)in_sizes; (void)n_in; (void)out_size; (void)ws_size;
  const float* x_l   = (const float*)d_in[0];
  const float* x_r   = (const float*)d_in[1];
  const float* d_l   = (const float*)d_in[2];
  const float* d_r   = (const float*)d_in[3];
  const float* gamma = (const float*)d_in[4];
  const float* beta  = (const float*)d_in[5];
  const float* w1    = (const float*)d_in[6];
  const float* b1    = (const float*)d_in[7];
  const float* w2    = (const float*)d_in[8];
  const float* b2    = (const float*)d_in[9];
  const float* qw    = (const float*)d_in[10];
  const float* qb    = (const float*)d_in[11];

  char* ws = (char*)d_ws;
  auto carve = [&](size_t bytes) -> char* {
    char* p = ws; ws += (bytes + 255) & ~(size_t)255; return p;
  };
  float* stats = (float*)carve(256*sizeof(float));
  float* y_l = (float*)carve((size_t)N1*4);
  float* y_r = (float*)carve((size_t)N1*4);
  _Float16* yh_l = (_Float16*)carve((size_t)N1*2);
  _Float16* yh_r = (_Float16*)carve((size_t)N1*2);
  _Float16* th_l = (_Float16*)carve((size_t)N1*2);
  _Float16* th_r = (_Float16*)carve((size_t)N1*2);
  _Float16* rh_l = (_Float16*)carve((size_t)N1*2);
  _Float16* rh_r = (_Float16*)carve((size_t)N1*2);
  _Float16* Qh_l = (_Float16*)carve((size_t)N1*2);
  _Float16* Kh_l = (_Float16*)carve((size_t)N1*2);
  _Float16* Vh_l = (_Float16*)carve((size_t)N1*2);
  _Float16* Qh_r = (_Float16*)carve((size_t)N1*2);
  _Float16* Kh_r = (_Float16*)carve((size_t)N1*2);
  _Float16* Vh_r = (_Float16*)carve((size_t)N1*2);
  float* M0 = (float*)carve((size_t)NWIN*4096*4);  // Mr2l
  float* M1 = (float*)carve((size_t)NWIN*4096*4);  // Ml2r

  bn_stats_kernel<<<128, 256, 0, stream>>>(x_l, x_r, stats);
  bn_apply_kernel<<<(2*N1)/256, 256, 0, stream>>>(x_l, x_r, gamma, beta, stats,
                                                  y_l, y_r, yh_l, yh_r);

  dim3 cg(WW/16, HH/8, BN*GG);
  gconv3_wmma_kernel<<<cg, 256, 0, stream>>>(yh_l, w1, b1, nullptr, th_l, 1);
  gconv3_wmma_kernel<<<cg, 256, 0, stream>>>(yh_r, w1, b1, nullptr, th_r, 1);
  gconv3_wmma_kernel<<<cg, 256, 0, stream>>>(th_l, w2, b2, y_l, rh_l, 0);
  gconv3_wmma_kernel<<<cg, 256, 0, stream>>>(th_r, w2, b2, y_r, rh_r, 0);

  qkv_wmma_kernel<<<(BN*HWP)/128, 256, 0, stream>>>(rh_l, qw, qb, Qh_l, Kh_l, Vh_l);
  qkv_wmma_kernel<<<(BN*HWP)/128, 256, 0, stream>>>(rh_r, qw, qb, Qh_r, Kh_r, Vh_r);

  // score_r2l: Q(left) x gathered K(right), idx = w - d_left
  attn_scores_kernel<<<NWIN, 256, 0, stream>>>(Qh_l, Kh_r, d_l, -1, M0);
  // score_l2r: Q(right) x gathered K(left), idx = w + d_right
  attn_scores_kernel<<<NWIN, 256, 0, stream>>>(Qh_r, Kh_l, d_r, +1, M1);

  attn_out_kernel<<<NWIN, 256, 0, stream>>>(M0, M1, Vh_l, Vh_r, d_l, d_r, x_l, x_r,
                                            (float*)d_out);
}
